// Attention_86646670230044
// MI455X (gfx1250) — matrix operand order repaired and checked by reading, not compiled, not run on hardware
//
#include <hip/hip_runtime.h>
#include <hip/hip_bf16.h>

typedef unsigned int   u32;
typedef unsigned short u16;

typedef __attribute__((ext_vector_type(16))) __bf16 v16bf;
typedef __attribute__((ext_vector_type(8)))  float  v8f;

union FragBF {            // 16 bf16 elements == 8 VGPRs, matches WMMA A/B operand
    v16bf v;
    u32   u[8];
};
union U4U16 {
    uint4 v;
    u16   s[8];
};

__device__ __forceinline__ u16 f32_to_bf16(float f) {
    u32 u = __float_as_uint(f);
    u += 0x7FFFu + ((u >> 16) & 1u);   // round-to-nearest-even
    return (u16)(u >> 16);
}
__device__ __forceinline__ float bf16_to_f32(u16 h) {
    return __uint_as_float(((u32)h) << 16);
}
__device__ __forceinline__ u32 pack2_bf16(float lo, float hi) {
    return (u32)f32_to_bf16(lo) | ((u32)f32_to_bf16(hi) << 16);
}

// ---------------------------------------------------------------------------
// CDNA5 async global->LDS copies (ASYNCcnt path). Guarded so the source still
// compiles (regular b128 copy fallback) if this toolchain lacks the builtins.
// Builtin signature (probe-confirmed via hipcc diagnostic): (V4i addrspace(1)*,
// V4i addrspace(3)*, imm offset, imm cpol).
// ---------------------------------------------------------------------------
#if defined(__has_builtin)
#if __has_builtin(__builtin_amdgcn_global_load_async_to_lds_b128) && \
    __has_builtin(__builtin_amdgcn_s_wait_asynccnt)
#define HAVE_ASYNC_LDS 1
#endif
#endif
#ifndef HAVE_ASYNC_LDS
#define HAVE_ASYNC_LDS 0
#endif

typedef int v4i_vec __attribute__((vector_size(16)));
typedef __attribute__((address_space(1))) v4i_vec glb_v4i;
typedef __attribute__((address_space(3))) v4i_vec lds_v4i;

__device__ __forceinline__ void copy_b128_to_lds(u16* lds_dst, const u16* gsrc) {
#if HAVE_ASYNC_LDS
    // generic->AS3: LDS offset lives in the low 32 bits of the generic address
    __builtin_amdgcn_global_load_async_to_lds_b128(
        (glb_v4i*)(uintptr_t)gsrc,
        (lds_v4i*)(u32)(uintptr_t)lds_dst,
        0, 0);
#else
    *(uint4*)lds_dst = *(const uint4*)gsrc;
#endif
}
__device__ __forceinline__ void wait_async_copies() {
#if HAVE_ASYNC_LDS
    __builtin_amdgcn_s_wait_asynccnt(0);
#endif
}

// ---------------------------------------------------------------------------
// fp32 -> bf16 conversion, 4 elements/thread
// ---------------------------------------------------------------------------
__global__ void cvt_f32_bf16_v4(const float4* __restrict__ in, uint2* __restrict__ out, int n4) {
    int i = blockIdx.x * 256 + threadIdx.x;
    if (i < n4) {
        float4 f = in[i];
        uint2 o;
        o.x = pack2_bf16(f.x, f.y);
        o.y = pack2_bf16(f.z, f.w);
        out[i] = o;
    }
}

// ---------------------------------------------------------------------------
// GEMM: Out[M,N] = A[M,K] @ W[N,K]^T   (A, W bf16; accumulate f32)
// Block tile 128x128, BK=32, double-buffered LDS with async copies:
// one block barrier per K-step; stage i+1 copies overlap stage i WMMAs.
// 8 waves: 4(M) x 2(N); each wave owns 32x64 = 2x4 tiles of 16x16.
// ---------------------------------------------------------------------------
template <bool OUT_F32>
__global__ __launch_bounds__(256) void gemm_bf16_wmma(
    const u16* __restrict__ A, const u16* __restrict__ W, void* __restrict__ Out,
    int M, int N, int K)
{
    constexpr int BM = 128, BN = 128, BK = 32;
    __shared__ __align__(16) u16 As[2][BM * BK];   // [row][k]
    __shared__ __align__(16) u16 Bs[2][BN * BK];   // [n][k]

    const int tid   = threadIdx.x;
    const int lane  = tid & 31;
    const int wave  = tid >> 5;
    const int waveM = wave >> 1;          // 0..3
    const int waveN = wave & 1;           // 0..1
    const int lm    = lane & 15;
    const int hi    = lane >> 4;          // half-wave select (ISA lane split)
    const size_t rowBase = (size_t)blockIdx.y * BM;
    const size_t colBase = (size_t)blockIdx.x * BN;

    v8f acc[2][4] = {};

    auto stage_load = [&](int buf, int k0) {
#pragma unroll
        for (int s = tid; s < BM * BK / 8; s += 256) {
            int row = s >> 2, c = s & 3;
            copy_b128_to_lds(&As[buf][s * 8], A + (rowBase + row) * (size_t)K + k0 + c * 8);
        }
#pragma unroll
        for (int s = tid; s < BN * BK / 8; s += 256) {
            int row = s >> 2, c = s & 3;
            copy_b128_to_lds(&Bs[buf][s * 8], W + (colBase + row) * (size_t)K + k0 + c * 8);
        }
    };

    stage_load(0, 0);
    int cur = 0;
    for (int k0 = 0; k0 < K; k0 += BK) {
        wait_async_copies();               // my stage-cur copies have landed
        __syncthreads();                   // everyone's landed; prev compute done
        if (k0 + BK < K) stage_load(cur ^ 1, k0 + BK);   // overlap with compute

        // -------- build fragments per ISA 7.12.2 layouts ----------------
        FragBF afr[2], bfr[4];
#pragma unroll
        for (int tm = 0; tm < 2; ++tm) {
            const u16* ar = &As[cur][(waveM * 32 + tm * 16 + lm) * BK];
#pragma unroll
            for (int i = 0; i < 4; ++i) {
                int kk = 2 * i + 8 * hi;                        // A: VGPR0-3 K=0..7(+8 hi)
                afr[tm].u[i]     = *(const u32*)(ar + kk);
                afr[tm].u[4 + i] = *(const u32*)(ar + 16 + kk); // VGPR4-7 K=16..23(+8 hi)
            }
        }
#pragma unroll
        for (int tn = 0; tn < 4; ++tn) {
            const u16* br = &Bs[cur][(waveN * 64 + tn * 16 + lm) * BK];
#pragma unroll
            for (int i = 0; i < 8; ++i)                         // B: VGPR0-3 K=0..7, VGPR4-7 K=16..23 (+8 hi)
                bfr[tn].u[i] = *(const u32*)(br + 2 * (i & 3) + 16 * (i >> 2) + 8 * hi);
        }
#pragma unroll
        for (int tm = 0; tm < 2; ++tm)
#pragma unroll
            for (int tn = 0; tn < 4; ++tn)
                acc[tm][tn] = __builtin_amdgcn_wmma_f32_16x16x32_bf16(
                    false, afr[tm].v, false, bfr[tn].v, (short)0, acc[tm][tn], false, false);
        cur ^= 1;
    }

    // -------- epilogue: D layout lane->(M=r+8*hi, N=lm) ------------------
#pragma unroll
    for (int tm = 0; tm < 2; ++tm)
#pragma unroll
        for (int tn = 0; tn < 4; ++tn) {
            size_t col = colBase + waveN * 64 + tn * 16 + lm;
#pragma unroll
            for (int r = 0; r < 8; ++r) {
                size_t row = rowBase + waveM * 32 + tm * 16 + r + 8 * hi;
                float v = acc[tm][tn][r];
                if (OUT_F32) ((float*)Out)[row * N + col] = v;
                else         ((u16*)Out)[row * N + col]   = f32_to_bf16(v);
            }
        }
}

// ---------------------------------------------------------------------------
// RoPE on bf16 Q or K, interleaved-pair convention, theta = 5e5, HD = 128
// ---------------------------------------------------------------------------
__global__ void rope_bf16(u16* __restrict__ qk, int S, int H, int W,
                          const int* __restrict__ start_pos)
{
    int idx = blockIdx.x * 256 + threadIdx.x;
    int ppr = H * 64;                       // pairs per sequence row
    if (idx >= S * ppr) return;
    int s = idx / ppr, rem = idx % ppr;
    int h = rem >> 6, d = rem & 63;
    const float LN_THETA = 13.122363377404328f;   // ln(500000)
    float freq = __expf(-((float)(2 * d) * (1.0f / 128.0f)) * LN_THETA);
    float ang  = (float)(s + *start_pos) * freq;
    float c = __cosf(ang), sn = __sinf(ang);
    u32* p = (u32*)(qk + (size_t)s * W + h * 128 + 2 * d);
    u32 pv = *p;
    float x0 = bf16_to_f32((u16)(pv & 0xFFFF));
    float x1 = bf16_to_f32((u16)(pv >> 16));
    *p = pack2_bf16(x0 * c - x1 * sn, x0 * sn + x1 * c);
}

// ---------------------------------------------------------------------------
// Flash attention: block = (head, 128-query tile), 8 waves x 16 rows.
// QK^T and P*V via v_wmma_f32_16x16x32_bf16, fp32 online softmax.
// Two block barriers per 64-key tile; per-wave P round-trip uses in-order DS
// ops + wave_barrier. B/V fragments rotate through two register buffers so
// each WMMA overlaps the next fragment's LDS loads (partial dscnt waits).
// ---------------------------------------------------------------------------
__global__ __launch_bounds__(256) void attn_wmma(
    const u16* __restrict__ Q,   // [S,4096] bf16 (post-RoPE)
    const u16* __restrict__ Kb,  // [S,1024] bf16 (post-RoPE)
    const u16* __restrict__ Vb,  // [S,1024] bf16
    u16* __restrict__ Ctx,       // [S,4096] bf16
    const int* __restrict__ start_pos)
{
    constexpr int HD = 128, QW = 4096, KVW = 1024, BQ = 128, BKV = 64;
    __shared__ __align__(16) u16 Ks[BKV * HD];    // [key][d]   16 KB
    __shared__ __align__(16) u16 Vt[HD * BKV];    // [d][key]   16 KB (transposed)
    __shared__ __align__(16) u16 Ps[8 * 16 * 32]; // per-wave P scratch, 8 KB

    const int h    = blockIdx.x;        // 0..31
    const int qblk = blockIdx.y;        // 0..15
    const int hk   = h >> 2;            // GQA: 4 Q-heads per KV-head
    const int tid  = threadIdx.x, lane = tid & 31, wave = tid >> 5;
    const int lm   = lane & 15, hi = lane >> 4;
    const int q0   = qblk * BQ;
    const int qrow0 = q0 + wave * 16;
    const int sp   = *start_pos;
    const float scale = 0.08838834764831845f;    // 1/sqrt(128)

    // ---- this wave's 16x128 Q tile as 4 A-fragments (2 x b128 each) ------
    FragBF qfr[4];
    {
        const u16* qr = Q + (size_t)(qrow0 + lm) * QW + h * HD;
#pragma unroll
        for (int kc = 0; kc < 4; ++kc) {
            uint4 a = *(const uint4*)(qr + kc * 32 + 16 * hi);      // K = 0..7 (+16 hi)
            uint4 b = *(const uint4*)(qr + kc * 32 + 8 + 16 * hi);  // K = 8..15 (+16 hi)
            qfr[kc].u[0] = a.x; qfr[kc].u[1] = a.y; qfr[kc].u[2] = a.z; qfr[kc].u[3] = a.w;
            qfr[kc].u[4] = b.x; qfr[kc].u[5] = b.y; qfr[kc].u[6] = b.z; qfr[kc].u[7] = b.w;
        }
    }

    v8f oacc[8] = {};                    // 16x128 output accumulator (8 d-tiles)
    float mrow[8], lrow[8];
#pragma unroll
    for (int r = 0; r < 8; ++r) { mrow[r] = -3.0e38f; lrow[r] = 0.0f; }

    const int kend = (qblk + 1) * BQ;    // causal: keys < q0+128
    for (int j0 = 0; j0 < kend; j0 += BKV) {
        __syncthreads();                 // previous tile's readers are done
        // ---- K tile: async global->LDS --------------------------------
#pragma unroll
        for (int t = tid; t < BKV * HD / 8; t += 256) {
            int key = t >> 4, c = t & 15;
            copy_b128_to_lds(&Ks[t * 8], Kb + (size_t)(j0 + key) * KVW + hk * HD + c * 8);
        }
        // ---- V tile: load + transpose into Vt[d][key] ------------------
#pragma unroll
        for (int t = tid; t < BKV * HD / 8; t += 256) {
            int key = t >> 4, c = t & 15;
            U4U16 u; u.v = *(const uint4*)(Vb + (size_t)(j0 + key) * KVW + hk * HD + c * 8);
#pragma unroll
            for (int e = 0; e < 8; ++e) Vt[(c * 8 + e) * BKV + key] = u.s[e];
        }
        wait_async_copies();
        __syncthreads();

        for (int jj = 0; jj < BKV; jj += 32) {       // 32 keys per inner step
            // ---- scores: S[16q x 32k] as two 16x16 D tiles --------------
            // (kt,kc) steps: (0,0)(1,0)(0,1)(1,1)... with B-fragment
            // double-buffering so WMMA overlaps the next fragment's loads.
            const u16* kr0 = Ks + (jj + lm) * HD;
            const u16* kr1 = Ks + (jj + 16 + lm) * HD;
            v8f st[2] = {};
            FragBF bfb[2];
#pragma unroll
            for (int i = 0; i < 8; ++i)
                bfb[0].u[i] = *(const u32*)(kr0 + 2 * i + 16 * hi);
#pragma unroll
            for (int step = 0; step < 8; ++step) {
                int kt = step & 1, kc = step >> 1;
                if (step < 7) {
                    int ns = step + 1;
                    const u16* kr = (ns & 1) ? kr1 : kr0;
                    int nkc = ns >> 1;
#pragma unroll
                    for (int i = 0; i < 8; ++i)
                        bfb[ns & 1].u[i] = *(const u32*)(kr + nkc * 32 + 2 * i + 16 * hi);
                }
                st[kt] = __builtin_amdgcn_wmma_f32_16x16x32_bf16(
                    false, qfr[kc].v, false, bfb[step & 1].v, (short)0, st[kt], false, false);
            }
            // ---- scale + causal mask + row max --------------------------
            float msub[8];
#pragma unroll
            for (int r = 0; r < 8; ++r) {
                int qi  = qrow0 + r + 8 * hi + sp;
                int kj0 = j0 + jj + lm;
                float s0 = st[0][r] * scale; if (kj0      > qi) s0 = -3.0e38f;
                float s1 = st[1][r] * scale; if (kj0 + 16 > qi) s1 = -3.0e38f;
                st[0][r] = s0; st[1][r] = s1;
                msub[r] = fmaxf(s0, s1);
            }
#pragma unroll
            for (int off = 1; off < 16; off <<= 1)
#pragma unroll
                for (int r = 0; r < 8; ++r)
                    msub[r] = fmaxf(msub[r], __shfl_xor(msub[r], off, 32));
            // ---- online softmax update (lane-local per row) -------------
            float psum[8];
#pragma unroll
            for (int r = 0; r < 8; ++r) {
                float mnew  = fmaxf(mrow[r], msub[r]);
                float alpha = __expf(mrow[r] - mnew);
                mrow[r] = mnew;
                lrow[r] *= alpha;
#pragma unroll
                for (int dt = 0; dt < 8; ++dt) oacc[dt][r] *= alpha;
                float p0 = __expf(st[0][r] - mnew);
                float p1 = __expf(st[1][r] - mnew);
                st[0][r] = p0; st[1][r] = p1;
                psum[r] = p0 + p1;
            }
#pragma unroll
            for (int off = 1; off < 16; off <<= 1)
#pragma unroll
                for (int r = 0; r < 8; ++r)
                    psum[r] += __shfl_xor(psum[r], off, 32);
#pragma unroll
            for (int r = 0; r < 8; ++r) lrow[r] += psum[r];

            // ---- D-layout -> A-layout via per-wave LDS scratch ----------
            // Intra-wave only: DS ops are in-order per wave; wave_barrier
            // just pins the compiler's schedule.
            u16* pw = Ps + wave * 16 * 32;
#pragma unroll
            for (int r = 0; r < 8; ++r) {
                int row = r + 8 * hi;
                pw[row * 32 + lm]      = f32_to_bf16(st[0][r]);
                pw[row * 32 + 16 + lm] = f32_to_bf16(st[1][r]);
            }
            __builtin_amdgcn_wave_barrier();
            FragBF pf;
            {
                const u16* pr = Ps + wave * 16 * 32 + lm * 32;
#pragma unroll
                for (int i = 0; i < 4; ++i) {
                    int kk = 2 * i + 16 * hi;
                    pf.u[i]     = *(const u32*)(pr + kk);
                    pf.u[4 + i] = *(const u32*)(pr + 8 + kk);
                }
            }
            __builtin_amdgcn_wave_barrier();
            // ---- O += P(16x32) @ V(32x128), V-fragment double-buffered --
            FragBF vfb[2];
            {
                const u16* vr = Vt + lm * BKV + jj;
#pragma unroll
                for (int i = 0; i < 8; ++i)
                    vfb[0].u[i] = *(const u32*)(vr + 2 * i + 16 * hi);
            }
#pragma unroll
            for (int dt = 0; dt < 8; ++dt) {
                if (dt < 7) {
                    const u16* vr = Vt + ((dt + 1) * 16 + lm) * BKV + jj;
#pragma unroll
                    for (int i = 0; i < 8; ++i)
                        vfb[(dt + 1) & 1].u[i] = *(const u32*)(vr + 2 * i + 16 * hi);
                }
                oacc[dt] = __builtin_amdgcn_wmma_f32_16x16x32_bf16(
                    false, pf.v, false, vfb[dt & 1].v, (short)0, oacc[dt], false, false);
            }
        }
    }
    // ---- normalize + store context (bf16) -------------------------------
#pragma unroll
    for (int dt = 0; dt < 8; ++dt) {
        size_t col = (size_t)h * HD + dt * 16 + lm;
#pragma unroll
        for (int r = 0; r < 8; ++r) {
            size_t row = qrow0 + r + 8 * hi;
            Ctx[row * QW + col] = f32_to_bf16(oacc[dt][r] / lrow[r]);
        }
    }
}

// ---------------------------------------------------------------------------
extern "C" void kernel_launch(void* const* d_in, const int* in_sizes, int n_in,
                              void* d_out, int out_size, void* d_ws, size_t ws_size,
                              hipStream_t stream)
{
    constexpr int S = 2048, HID = 4096, KVD = 1024;
    const float* x  = (const float*)d_in[0];
    const float* wq = (const float*)d_in[1];
    const float* wk = (const float*)d_in[2];
    const float* wv = (const float*)d_in[3];
    const float* wo = (const float*)d_in[4];
    const int*   sp = (const int*)d_in[5];

    u16* ws = (u16*)d_ws;
    size_t off = 0;
    u16* xb  = ws + off; off += (size_t)S * HID;       // 8 M
    u16* wqb = ws + off; off += (size_t)HID * HID;     // 16 M
    u16* wkb = ws + off; off += (size_t)KVD * HID;     // 4 M
    u16* wvb = ws + off; off += (size_t)KVD * HID;     // 4 M
    u16* wob = ws + off; off += (size_t)HID * HID;     // 16 M
    u16* qb  = ws + off; off += (size_t)S * HID;       // 8 M
    u16* kb  = ws + off; off += (size_t)S * KVD;       // 2 M
    u16* vb  = ws + off; off += (size_t)S * KVD;       // 2 M
    u16* ctx = ws + off; off += (size_t)S * HID;       // 8 M

    auto cvt = [&](const float* in, u16* out, int n) {
        cvt_f32_bf16_v4<<<(n / 4 + 255) / 256, 256, 0, stream>>>(
            (const float4*)in, (uint2*)out, n / 4);
    };
    cvt(x,  xb,  S * HID);
    cvt(wq, wqb, HID * HID);
    cvt(wk, wkb, KVD * HID);
    cvt(wv, wvb, KVD * HID);
    cvt(wo, wob, HID * HID);

    // Q/K/V projections (bf16 out)
    gemm_bf16_wmma<false><<<dim3(HID / 128, S / 128), 256, 0, stream>>>(xb, wqb, qb, S, HID, HID);
    gemm_bf16_wmma<false><<<dim3(KVD / 128, S / 128), 256, 0, stream>>>(xb, wkb, kb, S, KVD, HID);
    gemm_bf16_wmma<false><<<dim3(KVD / 128, S / 128), 256, 0, stream>>>(xb, wvb, vb, S, KVD, HID);

    // RoPE on Q and K
    rope_bf16<<<(S * 32 * 64 + 255) / 256, 256, 0, stream>>>(qb, S, 32, HID, sp);
    rope_bf16<<<(S *  8 * 64 + 255) / 256, 256, 0, stream>>>(kb, S,  8, KVD, sp);

    // Flash attention: grid (heads, query-tiles)
    attn_wmma<<<dim3(32, S / 128), 256, 0, stream>>>(qb, kb, vb, ctx, sp);

    // Output projection (f32 out -> d_out)
    gemm_bf16_wmma<true><<<dim3(HID / 128, S / 128), 256, 0, stream>>>(ctx, wob, d_out, S, HID, HID);
}
